// MultiHeadAttention_43121471651927
// MI455X (gfx1250) — matrix operand order repaired and checked by reading, not compiled, not run on hardware
//
#include <hip/hip_runtime.h>
#include <hip/hip_bf16.h>

// Problem constants (match reference)
#define BB 2
#define SS 4096
#define DD 512
#define HH 8
#define DKK 64
#define LDW 72   // padded LDS row stride in halves (144B: avoids bank conflicts)

typedef __attribute__((ext_vector_type(16))) _Float16 v16h;
typedef __attribute__((ext_vector_type(8)))  _Float16 v8h;
typedef __attribute__((ext_vector_type(8)))  float    v8f;

#if __has_builtin(__builtin_amdgcn_global_load_async_to_lds_b128) && \
    __has_builtin(__builtin_amdgcn_s_wait_asynccnt)
#define ASYNC_OK 1
typedef int v4i __attribute__((vector_size(16)));
typedef __attribute__((address_space(1))) v4i* gp128;   // global (AS1) int4*
typedef __attribute__((address_space(3))) v4i* lp128;   // LDS (AS3) int4*

// SFINAE probe: does the builtin accept this (global, lds) pointer-type pair?
template <typename...> struct mkvoid { typedef void type; };
template <typename G, typename L, typename = void>
struct has_async { static constexpr bool value = false; };
template <typename G, typename L>
struct has_async<G, L, typename mkvoid<decltype(
    __builtin_amdgcn_global_load_async_to_lds_b128((G)0, (L)0, 0, 0))>::type> {
  static constexpr bool value = true;
};

// LDS pointer materialization. AS3 value = low 32 bits of the generic pointer
// (ISA 10.2: flat->LDS aperture mapping truncates to addr[31:0]).
template <typename LP>
static __device__ __forceinline__ LP to_lp(_Float16* l) {
  if constexpr (__is_same(LP, lp128))
    return (LP)(unsigned int)(unsigned long long)l;
  else
    return (LP)l;
}

// Dependent-argument issue site: type-checked only on instantiation; the
// async branch is discarded (never instantiated) when the probe fails.
template <typename GP, typename LP>
static __device__ __forceinline__ void cp16_impl(const _Float16* g, _Float16* l) {
  if constexpr (has_async<GP, LP>::value) {
    __builtin_amdgcn_global_load_async_to_lds_b128(
        (GP)(unsigned long long)g, to_lp<LP>(l), 0, 0);
  } else {
    *(v8h*)l = *(const v8h*)g;
  }
}
#endif

// ---- 16B global -> LDS copy: async DMA if available, else load+store ----
static __device__ __forceinline__ void cp16(const _Float16* g, _Float16* l) {
#if ASYNC_OK
  if constexpr (has_async<gp128, lp128>::value) { cp16_impl<gp128, lp128>(g, l); return; }
  else if constexpr (has_async<gp128, v4i*>::value) { cp16_impl<gp128, v4i*>(g, l); return; }
  else if constexpr (has_async<v4i*, v4i*>::value) { cp16_impl<v4i*, v4i*>(g, l); return; }
#endif
  *(v8h*)l = *(const v8h*)g;
}

static __device__ __forceinline__ bool async_active() {
#if ASYNC_OK
  return has_async<gp128, lp128>::value || has_async<gp128, v4i*>::value ||
         has_async<v4i*, v4i*>::value;
#else
  return false;
#endif
}

static __device__ __forceinline__ void wait_async() {
#if ASYNC_OK
  if (async_active()) __builtin_amdgcn_s_wait_asynccnt(0);
#endif
}

// ---------------------------------------------------------------------------
// WMMA fragment loaders (wave32, V_WMMA_F32_16X16X32_F16 layouts per CDNA5 ISA)
//
// A (16x32 f16, MxK): lane<16 -> M=lane, halves[0..7]=K 0..7, halves[8..15]=K 16..23
//                     lane>=16 -> M=lane-16, halves = K 8..15 and K 24..31
// B (32x16 f16, KxN): lane<16 -> N=lane,  halves[0..7]=K 0..7, halves[8..15]=K 16..23
//                     lane>=16 -> N=lane-16, halves = K 8..15 and K 24..31
// All B-side matrices are stored pre-transposed (row = N, col = K contiguous)
// so each lane's B fragment is two contiguous 8-half runs, K k0.. and k0+16.. (2x b128 loads).
// ---------------------------------------------------------------------------

static __device__ __forceinline__ v16h load_a_f16(const _Float16* p, int ld) {
  const int lane  = threadIdx.x & 31;
  const int m     = lane & 15;
  const int k0    = (lane & 16) ? 8 : 0;
  const _Float16* row = p + (size_t)m * ld + k0;
  v8h lo = *(const v8h*)(row);
  v8h hi = *(const v8h*)(row + 16);
  v16h out;
#pragma unroll
  for (int i = 0; i < 8; ++i) { out[i] = lo[i]; out[i + 8] = hi[i]; }
  return out;
}

static __device__ __forceinline__ v16h load_a_f32(const float* p, int ld) {
  const int lane  = threadIdx.x & 31;
  const int m     = lane & 15;
  const int k0    = (lane & 16) ? 8 : 0;
  const float* row = p + (size_t)m * ld + k0;
  v16h out;
#pragma unroll
  for (int i = 0; i < 8; ++i) out[i] = (_Float16)row[i];
#pragma unroll
  for (int i = 0; i < 8; ++i) out[i + 8] = (_Float16)row[16 + i];
  return out;
}

static __device__ __forceinline__ v16h load_bt_f16(const _Float16* p, int ld) {
  const int lane  = threadIdx.x & 31;
  const int n     = lane & 15;
  const int k0    = (lane & 16) ? 8 : 0;
  const _Float16* row = p + (size_t)n * ld + k0;
  v8h lo = *(const v8h*)(row);
  v8h hi = *(const v8h*)(row + 16);
  v16h out;
#pragma unroll
  for (int i = 0; i < 8; ++i) { out[i] = lo[i]; out[i + 8] = hi[i]; }
  return out;
}

static __device__ __forceinline__ v8f wmma16(v16h a, v16h b, v8f c) {
  return __builtin_amdgcn_wmma_f32_16x16x32_f16(
      /*neg_a=*/false, a, /*neg_b=*/false, b,
      /*c_mod=*/(short)0, c, /*reuse_a=*/false, /*reuse_b=*/false);
}

// ---------------------------------------------------------------------------
// Kernel 0: convert+transpose weights to f16. Wq gets the 1/sqrt(DK) scale.
// WT[h][dk][d]  (so B-fragments read contiguous K=d runs), WoT[n][k].
// ---------------------------------------------------------------------------
__global__ void k_wconv(const float* __restrict__ Wq, const float* __restrict__ Wk,
                        const float* __restrict__ Wv, const float* __restrict__ Wo,
                        _Float16* __restrict__ WqT, _Float16* __restrict__ WkT,
                        _Float16* __restrict__ WvT, _Float16* __restrict__ WoT) {
  const int WSZ = HH * DD * DKK;  // 262144
  int idx = blockIdx.x * blockDim.x + threadIdx.x;
  if (idx < 3 * WSZ) {
    int w = idx / WSZ, rem = idx % WSZ;
    int h = rem / (DD * DKK);
    int d = (rem / DKK) % DD;
    int k = rem % DKK;
    const float* src = (w == 0) ? Wq : (w == 1) ? Wk : Wv;
    _Float16*    dst = (w == 0) ? WqT : (w == 1) ? WkT : WvT;
    float v = src[rem];
    if (w == 0) v *= 0.125f;  // fold 1/sqrt(DK) into q
    dst[((size_t)h * DKK + k) * DD + d] = (_Float16)v;
  } else {
    int j = idx - 3 * WSZ;
    if (j < DD * DD) {
      int k = j / DD, n = j % DD;
      WoT[(size_t)n * DD + k] = (_Float16)Wo[j];
    }
  }
}

// ---------------------------------------------------------------------------
// Kernel 1: QKV projection. One wave -> 16 rows x 64 cols, K-loop over D.
// z=0 -> q (scaled), z=1 -> k, z=2 -> v stored TRANSPOSED as vT[bh][dk][s].
// ---------------------------------------------------------------------------
__global__ void __launch_bounds__(32) k_proj(
    const float* __restrict__ Q, const float* __restrict__ Kin,
    const float* __restrict__ V,
    const _Float16* __restrict__ WqT, const _Float16* __restrict__ WkT,
    const _Float16* __restrict__ WvT,
    _Float16* __restrict__ qh, _Float16* __restrict__ kh,
    _Float16* __restrict__ vT) {
  const int rt = blockIdx.x;       // 16-row tile of S
  const int bh = blockIdx.y;       // b*H + h
  const int pz = blockIdx.z;       // 0=q 1=k 2=v
  const int b = bh / HH, h = bh % HH;
  const int lane  = threadIdx.x & 31;
  const int nlan  = lane & 15;
  const int hioff = (lane & 16) ? 8 : 0;

  const float*    X = (pz == 0) ? Q : (pz == 1) ? Kin : V;
  const _Float16* W = (pz == 0) ? WqT : (pz == 1) ? WkT : WvT;

  const float*    xbase = X + ((size_t)b * SS + (size_t)rt * 16) * DD;
  const _Float16* wbase = W + (size_t)h * DKK * DD;

  v8f acc[4] = {};
  for (int kc = 0; kc < DD; kc += 32) {
    v16h a = load_a_f32(xbase + kc, DD);
    v16h bf[4];
#pragma unroll
    for (int nt = 0; nt < 4; ++nt)
      bf[nt] = load_bt_f16(wbase + (size_t)(nt * 16) * DD + kc, DD);
#pragma unroll
    for (int nt = 0; nt < 4; ++nt)
      acc[nt] = wmma16(a, bf[nt], acc[nt]);
  }

  if (pz < 2) {
    _Float16* dst = (pz == 0) ? qh : kh;
#pragma unroll
    for (int nt = 0; nt < 4; ++nt)
#pragma unroll
      for (int r = 0; r < 8; ++r) {
        int s = rt * 16 + r + hioff;
        dst[((size_t)bh * SS + s) * DKK + nt * 16 + nlan] = (_Float16)acc[nt][r];
      }
  } else {
#pragma unroll
    for (int nt = 0; nt < 4; ++nt)
#pragma unroll
      for (int r = 0; r < 8; ++r) {
        int s = rt * 16 + r + hioff;
        int f = nt * 16 + nlan;
        vT[((size_t)bh * DKK + f) * SS + s] = (_Float16)acc[nt][r];
      }
  }
}

// ---------------------------------------------------------------------------
// Kernel 2: causal flash attention. 4 waves/WG; wave = 16 query rows.
// K/V 64-key blocks staged global->LDS (async DMA when available),
// double-buffered: block j+1's DMA is issued right after the barrier and
// lands while block j is computed (8 score WMMA + softmax + 8 P@V WMMA).
// ---------------------------------------------------------------------------
__global__ void __launch_bounds__(128) k_attn(
    const _Float16* __restrict__ qh, const _Float16* __restrict__ kh,
    const _Float16* __restrict__ vT, _Float16* __restrict__ oc) {
  const int qt = blockIdx.x;           // 64-query tile
  const int bh = blockIdx.y;
  const int b = bh / HH, h = bh % HH;
  const int tid   = threadIdx.x;
  const int w     = tid >> 5;
  const int lane  = tid & 31;
  const int nlan  = lane & 15;
  const int hioff = (lane & 16) ? 8 : 0;
  const int qbase = qt * 64 + w * 16;

  __shared__ __align__(16) _Float16 kbuf[2][64 * LDW];
  __shared__ __align__(16) _Float16 vbuf[2][64 * LDW];
  __shared__ __align__(16) _Float16 pt[4][16 * LDW];
  _Float16* myp = &pt[w][0];

  const _Float16* kb0 = kh + (size_t)bh * SS * DKK;
  const _Float16* vb0 = vT + (size_t)bh * DKK * SS;

  // cooperative stage of one 64-key block: K tile 64x64 (row=key),
  // V tile 64x64 (row=feature, from vT with row stride SS). 512 16B chunks
  // each; 128 threads x 4 chunks.
  auto stage = [&](int j, int buf) {
    const _Float16* kg = kb0 + (size_t)j * 64 * DKK;
    const _Float16* vg = vb0 + (size_t)j * 64;
    _Float16* kl = &kbuf[buf][0];
    _Float16* vl = &vbuf[buf][0];
#pragma unroll
    for (int i = 0; i < 4; ++i) {
      int c = tid + 128 * i;
      int row = c >> 3, col = (c & 7) * 8;
      cp16(kg + (size_t)row * DKK + col, kl + row * LDW + col);
    }
#pragma unroll
    for (int i = 0; i < 4; ++i) {
      int c = tid + 128 * i;
      int row = c >> 3, col = (c & 7) * 8;
      cp16(vg + (size_t)row * SS + col, vl + row * LDW + col);
    }
  };

  const _Float16* qrow = qh + ((size_t)bh * SS + qbase) * DKK;
  v16h qa0 = load_a_f16(qrow, DKK);
  v16h qa1 = load_a_f16(qrow + 32, DKK);

  v8f o[4] = {};
  float m_i[8], l_i[8];
#pragma unroll
  for (int r = 0; r < 8; ++r) { m_i[r] = -1e30f; l_i[r] = 0.0f; }

  stage(0, 0);  // prologue DMA

  for (int j = 0; j <= qt; ++j) {
    const int buf = j & 1;
    wait_async();       // this wave's DMA for block j has landed
    __syncthreads();    // all waves' DMA landed; previous buffer free
    if (j < qt) stage(j + 1, buf ^ 1);  // overlap next DMA with compute

    const _Float16* kl = &kbuf[buf][0];
    const _Float16* vl = &vbuf[buf][0];

    // ---- scores: S = q @ k^T  (16 x 64), fragments from LDS ----
    v16h kf[8];
#pragma unroll
    for (int nt = 0; nt < 4; ++nt) {
      kf[2 * nt]     = load_bt_f16(kl + nt * 16 * LDW, LDW);
      kf[2 * nt + 1] = load_bt_f16(kl + nt * 16 * LDW + 32, LDW);
    }
    v8f s[4] = {};
#pragma unroll
    for (int nt = 0; nt < 4; ++nt) {
      s[nt] = wmma16(qa0, kf[2 * nt], s[nt]);
      s[nt] = wmma16(qa1, kf[2 * nt + 1], s[nt]);
    }

    // ---- V fragments (independent of softmax; LDS latency is short) ----
    v16h vf[8];
#pragma unroll
    for (int nt = 0; nt < 4; ++nt) {
      vf[2 * nt]     = load_bt_f16(vl + nt * 16 * LDW, LDW);
      vf[2 * nt + 1] = load_bt_f16(vl + nt * 16 * LDW + 32, LDW);
    }

    if (j == qt) {  // diagonal block: causal mask
#pragma unroll
      for (int nt = 0; nt < 4; ++nt)
#pragma unroll
        for (int r = 0; r < 8; ++r) {
          int sq = qbase + r + hioff;
          int kk = j * 64 + nt * 16 + nlan;
          if (kk > sq) s[nt][r] = -1e30f;
        }
    }

    // ---- online softmax (row stats replicated per 16-lane half) ----
    float bm[8];
#pragma unroll
    for (int r = 0; r < 8; ++r) {
      float v = fmaxf(fmaxf(s[0][r], s[1][r]), fmaxf(s[2][r], s[3][r]));
      v = fmaxf(v, __shfl_xor(v, 1, 32));
      v = fmaxf(v, __shfl_xor(v, 2, 32));
      v = fmaxf(v, __shfl_xor(v, 4, 32));
      v = fmaxf(v, __shfl_xor(v, 8, 32));
      bm[r] = v;
    }
    float alpha[8];
#pragma unroll
    for (int r = 0; r < 8; ++r) {
      float mn = fmaxf(m_i[r], bm[r]);
      alpha[r] = __expf(m_i[r] - mn);
      m_i[r] = mn;
    }
#pragma unroll
    for (int nt = 0; nt < 4; ++nt)
#pragma unroll
      for (int r = 0; r < 8; ++r) o[nt][r] *= alpha[r];

    float rs[8];
#pragma unroll
    for (int r = 0; r < 8; ++r) rs[r] = 0.0f;
#pragma unroll
    for (int nt = 0; nt < 4; ++nt)
#pragma unroll
      for (int r = 0; r < 8; ++r) {
        float p = __expf(s[nt][r] - m_i[r]);
        rs[r] += p;
        myp[(r + hioff) * LDW + nt * 16 + nlan] = (_Float16)p;
      }
#pragma unroll
    for (int r = 0; r < 8; ++r) {
      float v = rs[r];
      v += __shfl_xor(v, 1, 32);
      v += __shfl_xor(v, 2, 32);
      v += __shfl_xor(v, 4, 32);
      v += __shfl_xor(v, 8, 32);
      l_i[r] = l_i[r] * alpha[r] + v;
    }

    // ---- O += P @ V  (P re-read as A fragments from private LDS tile) ----
    v16h pa0 = load_a_f16(myp, LDW);
    v16h pa1 = load_a_f16(myp + 32, LDW);
#pragma unroll
    for (int nt = 0; nt < 4; ++nt) {
      o[nt] = wmma16(pa0, vf[2 * nt], o[nt]);
      o[nt] = wmma16(pa1, vf[2 * nt + 1], o[nt]);
    }
  }

  // ---- normalize and store concat-head layout oc[b][s][h*DK + f] ----
#pragma unroll
  for (int r = 0; r < 8; ++r) {
    float inv = 1.0f / l_i[r];
    int sq = qbase + r + hioff;
    _Float16* dst = oc + ((size_t)b * SS + sq) * (HH * DKK) + (size_t)h * DKK;
#pragma unroll
    for (int nt = 0; nt < 4; ++nt)
      dst[nt * 16 + nlan] = (_Float16)(o[nt][r] * inv);
  }
}

// ---------------------------------------------------------------------------
// Kernel 3: output projection  out = oc @ Wo  (fp32 out).
// One wave -> 16 rows x 64 cols.
// ---------------------------------------------------------------------------
__global__ void __launch_bounds__(32) k_out(
    const _Float16* __restrict__ oc, const _Float16* __restrict__ WoT,
    float* __restrict__ out) {
  const int rt = blockIdx.x;  // 16-row tile of B*S
  const int ct = blockIdx.y;  // 64-col tile of D
  const int lane  = threadIdx.x & 31;
  const int nlan  = lane & 15;
  const int hioff = (lane & 16) ? 8 : 0;

  const _Float16* xb = oc + (size_t)rt * 16 * DD;
  v8f acc[4] = {};
  for (int kc = 0; kc < DD; kc += 32) {
    v16h a = load_a_f16(xb + kc, DD);
    v16h bf[4];
#pragma unroll
    for (int nt = 0; nt < 4; ++nt)
      bf[nt] = load_bt_f16(WoT + (size_t)(ct * 64 + nt * 16) * DD + kc, DD);
#pragma unroll
    for (int nt = 0; nt < 4; ++nt)
      acc[nt] = wmma16(a, bf[nt], acc[nt]);
  }
#pragma unroll
  for (int nt = 0; nt < 4; ++nt)
#pragma unroll
    for (int r = 0; r < 8; ++r) {
      int row = rt * 16 + r + hioff;
      out[(size_t)row * DD + ct * 64 + nt * 16 + nlan] = acc[nt][r];
    }
}

// ---------------------------------------------------------------------------
extern "C" void kernel_launch(void* const* d_in, const int* in_sizes, int n_in,
                              void* d_out, int out_size, void* d_ws, size_t ws_size,
                              hipStream_t stream) {
  (void)in_sizes; (void)n_in; (void)out_size; (void)ws_size;
  const float* Q  = (const float*)d_in[0];
  const float* Kp = (const float*)d_in[1];
  const float* V  = (const float*)d_in[2];
  const float* Wq = (const float*)d_in[3];
  const float* Wk = (const float*)d_in[4];
  const float* Wv = (const float*)d_in[5];
  const float* Wo = (const float*)d_in[6];
  float* out = (float*)d_out;

  // Workspace layout (f16 elements); total ~34 MB.
  _Float16* ws = (_Float16*)d_ws;
  const size_t QKV = (size_t)BB * HH * SS * DKK;  // 4,194,304
  const size_t WSZ = (size_t)HH * DD * DKK;       //   262,144
  _Float16* qh  = ws;
  _Float16* kh  = qh + QKV;
  _Float16* vT  = kh + QKV;
  _Float16* oc  = vT + QKV;
  _Float16* WqT = oc + QKV;
  _Float16* WkT = WqT + WSZ;
  _Float16* WvT = WkT + WSZ;
  _Float16* WoT = WvT + WSZ;

  const int tot = 3 * (int)WSZ + DD * DD;  // 1,048,576
  k_wconv<<<dim3(tot / 256), dim3(256), 0, stream>>>(Wq, Wk, Wv, Wo,
                                                     WqT, WkT, WvT, WoT);
  k_proj<<<dim3(SS / 16, BB * HH, 3), dim3(32), 0, stream>>>(
      Q, Kp, V, WqT, WkT, WvT, qh, kh, vT);
  k_attn<<<dim3(SS / 64, BB * HH), dim3(128), 0, stream>>>(qh, kh, vT, oc);
  k_out<<<dim3(BB * SS / 16, DD / 64), dim3(32), 0, stream>>>(oc, WoT, out);
}
